// VectorQuantizer_34402688041324
// MI455X (gfx1250) — compile-verified, matrix operand order checked by reading, not run on hardware
//
#include <hip/hip_runtime.h>
#include <hip/hip_bf16.h>
#include <stdint.h>

// ---------------------------------------------------------------------------
// Problem constants (from reference): z [8,64,64,64] f32, emb [8192,64] f32
// ---------------------------------------------------------------------------
#define N_ROWS   32768      // 8*64*64 latent vectors
#define N_CODES  8192
#define KDIM     64
#define NELEM    2097152    // N_ROWS * KDIM
#define NTILES   512        // N_CODES / 16
#define LDS_STRIDE 72       // padded row stride (144 B) -> bank-friendly b128

typedef __attribute__((ext_vector_type(16))) __bf16 v16bf;
typedef __attribute__((ext_vector_type(8)))  float  v8f;

union BV {              // one WMMA 16-bit operand: 16 bf16 = 2 x 16-byte chunks
    v16bf v;
    uint4 q[2];
};

__device__ __forceinline__ unsigned short bf16_rne(float f) {
    unsigned int u = __float_as_uint(f);
    u += 0x7fffu + ((u >> 16) & 1u);
    return (unsigned short)(u >> 16);
}
__device__ __forceinline__ float bf16_to_f32(unsigned short h) {
    return __uint_as_float(((unsigned int)h) << 16);
}

__device__ __forceinline__ v8f wmma_bf16(const BV& A, const BV& B, v8f C) {
    // D = A(16x32 bf16) x B(32x16 bf16) + C(16x16 f32)
    return __builtin_amdgcn_wmma_f32_16x16x32_bf16(
        /*neg_a=*/false, A.v, /*neg_b=*/false, B.v,
        /*c_mod=*/(short)0, C, /*reuse_a=*/false, /*reuse_b=*/false);
}

// ---------------------------------------------------------------------------
// Kernel 1: codebook -> bf16 hi/lo split + exact f32 ||e||^2
// ---------------------------------------------------------------------------
__global__ __launch_bounds__(256) void vq_prep_emb(
    const float* __restrict__ emb,
    unsigned short* __restrict__ e_hi, unsigned short* __restrict__ e_lo,
    float* __restrict__ e2)
{
    int m = blockIdx.x * 256 + threadIdx.x;
    if (m >= N_CODES) return;
    const float* row = emb + (size_t)m * KDIM;
    float s = 0.f;
#pragma unroll 8
    for (int c = 0; c < KDIM; ++c) {
        float w = row[c];
        s += w * w;
        unsigned short h = bf16_rne(w);
        float r = w - bf16_to_f32(h);
        e_hi[(size_t)m * KDIM + c] = h;
        e_lo[(size_t)m * KDIM + c] = bf16_rne(r);
    }
    e2[m] = s;
}

// ---------------------------------------------------------------------------
// Kernel 2: z NCHW -> [N][64] row-major bf16 hi/lo (transpose + split)
// ---------------------------------------------------------------------------
__global__ __launch_bounds__(256) void vq_prep_z(
    const float* __restrict__ z,
    unsigned short* __restrict__ zh, unsigned short* __restrict__ zl)
{
    int o = blockIdx.x * 256 + threadIdx.x;       // 0 .. NELEM-1 (n-major, c-minor)
    int n = o >> 6, c = o & 63;
    int b = n >> 12, hw = n & 4095;
    float v = z[((size_t)b << 18) + ((size_t)c << 12) + (size_t)hw];
    unsigned short h = bf16_rne(v);
    zh[o] = h;
    zl[o] = bf16_rne(v - bf16_to_f32(h));
}

// ---------------------------------------------------------------------------
// Kernel 3: distance GEMM + argmin.
//   block = 256 threads = 8 waves; wave owns rows [rb0, rb0+32) as two 16-row
//   WMMA tiles kept in registers (A operand, hi+lo, two K-tiles of 32).
//   Codebook tiles (16 codes, hi/lo + ||e||^2) double-buffered through LDS.
//   C is initialized with -||e||^2/2, so after the 6 split-precision WMMAs
//   acc = z.e - ||e||^2/2  and  argmin(d) == argmax(acc). No per-tile global
//   loads and no FMA tail in the critical path; one barrier per iteration.
// ---------------------------------------------------------------------------
__global__ __launch_bounds__(256) void vq_dist_argmin(
    const unsigned short* __restrict__ zh, const unsigned short* __restrict__ zl,
    const unsigned short* __restrict__ eh, const unsigned short* __restrict__ el,
    const float* __restrict__ e2g,
    int* __restrict__ idx_i, float* __restrict__ idx_f)
{
    __shared__ __align__(16) unsigned short lh[2][16 * LDS_STRIDE];
    __shared__ __align__(16) unsigned short ll[2][16 * LDS_STRIDE];
    __shared__ __align__(16) float          le2[2][16];

    const int tid  = threadIdx.x;
    const int lane = tid & 31;
    const int wave = tid >> 5;
    const int col  = lane & 15;                 // A row / B col / C col class
    const int K0   = (lane < 16) ? 0 : 8;       // 16-bit operand K-chunk base
    const int rb0  = blockIdx.x * 256 + wave * 32;

    // staging-role decomposition (256 threads move 256 x 16B per tile)
    const int scode = (tid & 127) >> 3;         // code row 0..15
    const int schnk = tid & 7;                  // 16B-chunk 0..7
    const bool s_lo = tid >= 128;

    // ---- load A operands: [row-tile][hi/lo][k-tile] ----
    BV a[2][2][2];
#pragma unroll
    for (int tt = 0; tt < 2; ++tt) {
        const int row = rb0 + tt * 16 + col;
        const unsigned short* ph = zh + (size_t)row * KDIM;
        const unsigned short* pl = zl + (size_t)row * KDIM;
#pragma unroll
        for (int kt = 0; kt < 2; ++kt) {
            a[tt][0][kt].q[0] = *(const uint4*)(ph + kt * 32 + K0);
            a[tt][0][kt].q[1] = *(const uint4*)(ph + kt * 32 + K0 + 16);
            a[tt][1][kt].q[0] = *(const uint4*)(pl + kt * 32 + K0);
            a[tt][1][kt].q[1] = *(const uint4*)(pl + kt * 32 + K0 + 16);
        }
    }

    float mv[2][8];                             // running max of z.e - ||e||^2/2
    int   mi[2][8];
#pragma unroll
    for (int tt = 0; tt < 2; ++tt)
#pragma unroll
        for (int i = 0; i < 8; ++i) { mv[tt][i] = -3.0e38f; mi[tt][i] = 0; }

    // ---- prologue: stage tile 0 into buffer 0 ----
    {
        const int c0 = 0;
        const unsigned short* src = (s_lo ? el : eh) + (size_t)(c0 + scode) * KDIM + schnk * 8;
        unsigned short* dst = (s_lo ? ll[0] : lh[0]) + scode * LDS_STRIDE + schnk * 8;
        *(uint4*)dst = *(const uint4*)src;
        if (tid < 16) le2[0][tid] = e2g[c0 + tid];
    }
    __syncthreads();

    for (int t = 0; t < NTILES; ++t) {
        const int buf = t & 1;

        // ---- stage next tile into the other buffer (overlaps with WMMAs) ----
        if (t + 1 < NTILES) {
            const int c1 = (t + 1) * 16;
            const unsigned short* src = (s_lo ? el : eh) + (size_t)(c1 + scode) * KDIM + schnk * 8;
            unsigned short* dst = (s_lo ? ll[buf ^ 1] : lh[buf ^ 1]) + scode * LDS_STRIDE + schnk * 8;
            *(uint4*)dst = *(const uint4*)src;
            if (tid < 16) le2[buf ^ 1][tid] = e2g[c1 + tid];
        }

        // ---- B operands from LDS (column = code col, K chunking as A) ----
        BV bh[2], bl[2];
        {
            const unsigned short* ph = lh[buf] + col * LDS_STRIDE;
            const unsigned short* pl = ll[buf] + col * LDS_STRIDE;
#pragma unroll
            for (int kt = 0; kt < 2; ++kt) {
                bh[kt].q[0] = *(const uint4*)(ph + kt * 32 + K0);
                bh[kt].q[1] = *(const uint4*)(ph + kt * 32 + K0 + 16);
                bl[kt].q[0] = *(const uint4*)(pl + kt * 32 + K0);
                bl[kt].q[1] = *(const uint4*)(pl + kt * 32 + K0 + 16);
            }
        }
        const float cinit = -0.5f * le2[buf][col];
        const int   code  = t * 16 + col;

#pragma unroll
        for (int tt = 0; tt < 2; ++tt) {
            v8f acc = {cinit, cinit, cinit, cinit, cinit, cinit, cinit, cinit};
            // split-precision: hi*hi + hi*lo + lo*hi, f32 accumulate
            acc = wmma_bf16(a[tt][0][0], bh[0], acc);
            acc = wmma_bf16(a[tt][0][0], bl[0], acc);
            acc = wmma_bf16(a[tt][1][0], bh[0], acc);
            acc = wmma_bf16(a[tt][0][1], bh[1], acc);
            acc = wmma_bf16(a[tt][0][1], bl[1], acc);
            acc = wmma_bf16(a[tt][1][1], bh[1], acc);
#pragma unroll
            for (int i = 0; i < 8; ++i) {
                bool better = acc[i] > mv[tt][i];   // strict > keeps first min
                mv[tt][i] = better ? acc[i] : mv[tt][i];
                mi[tt][i] = better ? code : mi[tt][i];
            }
        }
        __syncthreads();   // single barrier: ends tile t epoch, publishes t+1
    }

    // ---- cross-lane argmax within each 16-lane half (C-layout columns) ----
#pragma unroll
    for (int tt = 0; tt < 2; ++tt) {
#pragma unroll
        for (int i = 0; i < 8; ++i) {
            float v = mv[tt][i];
            int   ix = mi[tt][i];
#pragma unroll
            for (int m = 1; m < 16; m <<= 1) {
                float ov = __shfl_xor(v, m, 32);
                int   oi = __shfl_xor(ix, m, 32);
                bool take = (ov > v) || (ov == v && oi < ix);  // first-min tiebreak
                v  = take ? ov : v;
                ix = take ? oi : ix;
            }
            if (col == 0) {
                // VGPR i: lanes 0-15 -> row i, lanes 16-31 -> row i+8
                int r = rb0 + tt * 16 + i + ((lane >> 4) << 3);
                idx_i[r] = ix;
                idx_f[r] = (float)ix;
            }
        }
    }
}

// ---------------------------------------------------------------------------
// Kernel 4: z_q gather (NCHW) + per-block partial MSE sums (deterministic)
// ---------------------------------------------------------------------------
__global__ __launch_bounds__(256) void vq_zq_loss(
    const float* __restrict__ z, const float* __restrict__ emb,
    const int* __restrict__ idx,
    float* __restrict__ zq_out, float* __restrict__ partial)
{
    __shared__ float red[256];
    int o  = blockIdx.x * 256 + threadIdx.x;    // NCHW flat index
    int b  = o >> 18;
    int c  = (o >> 12) & 63;
    int hw = o & 4095;
    int n  = (b << 12) | hw;
    int m  = idx[n];
    float e  = emb[(size_t)m * KDIM + c];
    float zv = z[o];
    zq_out[o] = e;                              // straight-through == z_q numerically
    float d = e - zv;
    red[threadIdx.x] = d * d;
    __syncthreads();
#pragma unroll
    for (int s = 128; s > 0; s >>= 1) {
        if (threadIdx.x < s) red[threadIdx.x] += red[threadIdx.x + s];
        __syncthreads();
    }
    if (threadIdx.x == 0) partial[blockIdx.x] = red[0];
}

__global__ __launch_bounds__(256) void vq_loss_reduce(
    const float* __restrict__ partial, float* __restrict__ loss)
{
    __shared__ float red[256];
    float s = 0.f;
    for (int k = threadIdx.x; k < 8192; k += 256) s += partial[k];
    red[threadIdx.x] = s;
    __syncthreads();
#pragma unroll
    for (int st = 128; st > 0; st >>= 1) {
        if (threadIdx.x < st) red[threadIdx.x] += red[threadIdx.x + st];
        __syncthreads();
    }
    // loss = (1 + BETA) * mean((z_q - z)^2), BETA = 0.25
    if (threadIdx.x == 0) loss[0] = red[0] * (1.25f / 2097152.0f);
}

// ---------------------------------------------------------------------------
extern "C" void kernel_launch(void* const* d_in, const int* in_sizes, int n_in,
                              void* d_out, int out_size, void* d_ws, size_t ws_size,
                              hipStream_t stream)
{
    const float* z   = (const float*)d_in[0];   // [8,64,64,64]
    const float* emb = (const float*)d_in[1];   // [8192,64]
    float* out = (float*)d_out;                 // loss(1) + z_q(NELEM) + idx(N_ROWS)

    // workspace carve-up (~10.7 MB)
    unsigned short* e_hi = (unsigned short*)d_ws;
    unsigned short* e_lo = e_hi + (size_t)N_CODES * KDIM;
    unsigned short* z_hi = e_lo + (size_t)N_CODES * KDIM;
    unsigned short* z_lo = z_hi + (size_t)N_ROWS * KDIM;
    float* e2      = (float*)(z_lo + (size_t)N_ROWS * KDIM);
    int*   idx_i   = (int*)(e2 + N_CODES);
    float* partial = (float*)(idx_i + N_ROWS);

    float* loss  = out;
    float* zq    = out + 1;
    float* idx_f = out + 1 + NELEM;

    vq_prep_emb   <<<N_CODES / 256, 256, 0, stream>>>(emb, e_hi, e_lo, e2);
    vq_prep_z     <<<NELEM / 256,   256, 0, stream>>>(z, z_hi, z_lo);
    vq_dist_argmin<<<N_ROWS / 256,  256, 0, stream>>>(z_hi, z_lo, e_hi, e_lo, e2,
                                                      idx_i, idx_f);
    vq_zq_loss    <<<NELEM / 256,   256, 0, stream>>>(z, emb, idx_i, zq, partial);
    vq_loss_reduce<<<1,             256, 0, stream>>>(partial, loss);
}
